// GraphConvolutionWithEdgeConcat_38070590112205
// MI455X (gfx1250) — matrix operand order repaired and checked by reading, not compiled, not run on hardware
//
#include <hip/hip_runtime.h>

#define N_NODES 50000
#define NNZ_E   800000
#define R_REL   3
#define D       128
#define KTOT    512          // 3*128 (concat support) + 128 (layernorm branch)
#define EPSF    1e-6f

typedef __attribute__((ext_vector_type(16))) __bf16 v16bf;
typedef __attribute__((ext_vector_type(8)))  float  v8f;

union BF16x16 { v16bf v; unsigned int u[8]; };

__device__ __forceinline__ unsigned short f2bf(float f) {
  // round-to-nearest-even float -> bf16
  unsigned int u = __float_as_uint(f);
  u += 0x7FFFu + ((u >> 16) & 1u);
  return (unsigned short)(u >> 16);
}

// ---------------- zero the atomic accumulator ----------------
__global__ void k_zero_f32(float* __restrict__ p, int n4) {
  int i = blockIdx.x * blockDim.x + threadIdx.x;
  if (i < n4) ((float4*)p)[i] = make_float4(0.f, 0.f, 0.f, 0.f);
}

// ---------------- SpMM scatter: one wave per edge ----------------
// supp layout: [N][R*D], relation r occupies columns r*D .. r*D+127
__global__ void k_spmm(const float* __restrict__ x,
                       const int*   __restrict__ rows,
                       const int*   __restrict__ cols,
                       const float* __restrict__ vals,
                       float*       __restrict__ supp) {
  unsigned gid  = blockIdx.x * blockDim.x + threadIdx.x;
  unsigned e    = gid >> 5;        // edge id, flat over R*NNZ
  unsigned lane = gid & 31u;       // wave32 lane -> 4 dims each
  if (e >= (unsigned)(R_REL * NNZ_E)) return;
  unsigned r   = e / (unsigned)NNZ_E;
  int      row = rows[e];
  int      col = cols[e];
  float    v   = vals[e];
  float4 xi = ((const float4*)(x + (size_t)col * D))[lane];
  float* dst = supp + (size_t)row * (R_REL * D) + r * D + lane * 4;
  atomicAdd(dst + 0, v * xi.x);
  atomicAdd(dst + 1, v * xi.y);
  atomicAdd(dst + 2, v * xi.z);
  atomicAdd(dst + 3, v * xi.w);
}

// ---------------- layernorm(sum_r support_r): one wave per node ----------------
__global__ void k_layernorm(const float* __restrict__ supp,
                            const float* __restrict__ gamma,
                            const float* __restrict__ beta,
                            unsigned short* __restrict__ normbf) {
  unsigned wave = threadIdx.x >> 5;
  unsigned lane = threadIdx.x & 31u;
  unsigned node = blockIdx.x * 8u + wave;
  if (node >= (unsigned)N_NODES) return;
  const float4* sp = (const float4*)(supp + (size_t)node * (R_REL * D));
  float4 a = sp[lane];            // relation 0
  float4 b = sp[32 + lane];       // relation 1
  float4 c = sp[64 + lane];       // relation 2
  float4 s = make_float4(a.x + b.x + c.x, a.y + b.y + c.y,
                         a.z + b.z + c.z, a.w + b.w + c.w);
  float sum = s.x + s.y + s.z + s.w;
  float sq  = s.x * s.x + s.y * s.y + s.z * s.z + s.w * s.w;
  #pragma unroll
  for (int off = 16; off > 0; off >>= 1) {   // wave32 reduction
    sum += __shfl_xor(sum, off, 32);
    sq  += __shfl_xor(sq,  off, 32);
  }
  float mean = sum * (1.0f / D);
  float var  = sq  * (1.0f / D) - mean * mean;   // population var (jnp.var)
  float inv  = rsqrtf(var + EPSF);
  unsigned d0 = lane * 4u;
  float sv[4] = {s.x, s.y, s.z, s.w};
  unsigned short* o = normbf + (size_t)node * D + d0;
  #pragma unroll
  for (int j = 0; j < 4; ++j)
    o[j] = f2bf((sv[j] - mean) * inv * gamma[d0 + j] + beta[d0 + j]);
}

// ---------------- pack combined weight [W; Ws] -> bf16 [512][128] ----------------
__global__ void k_packw(const float* __restrict__ w,
                        const float* __restrict__ shw,
                        unsigned short* __restrict__ wc) {
  int i = blockIdx.x * blockDim.x + threadIdx.x;
  if (i >= KTOT * D) return;
  int row = i >> 7;
  int col = i & 127;
  float f = (row < R_REL * D) ? w[row * D + col] : shw[(row - R_REL * D) * D + col];
  wc[i] = f2bf(f);
}

// ---------------- WMMA GEMM: out = 0.5 * (A[N,512] @ Wc[512,128]) + bias ----------------
// Block = 8 waves, 16 rows; wave w computes the 16x16 tile at cols w*16.
__global__ void __launch_bounds__(256) k_gemm(
    const float* __restrict__ supp,            // [N][384] f32
    const unsigned short* __restrict__ normbf, // [N][128] bf16
    const unsigned short* __restrict__ wc,     // [512][128] bf16
    const float* __restrict__ bias,            // [128]
    float* __restrict__ out) {                 // [N][128] f32
  __shared__ unsigned short As[16 * KTOT];     // 16 KB A panel (bf16)
  const unsigned tid = threadIdx.x;
  const unsigned r0  = blockIdx.x * 16u;

  // stage A panel: cols 0..383 = support (f32 -> bf16), cols 384..511 = layernorm branch
  for (unsigned i = tid; i < 16u * 384u; i += 256u) {
    unsigned row = i / 384u, col = i - row * 384u;
    As[row * KTOT + col] = f2bf(supp[(size_t)(r0 + row) * 384u + col]);
  }
  for (unsigned i = tid; i < 16u * 128u; i += 256u) {
    unsigned row = i >> 7, col = i & 127u;
    As[row * KTOT + 384u + col] = normbf[(size_t)(r0 + row) * 128u + col];
  }
  __syncthreads();

  const unsigned wave = tid >> 5;
  const unsigned lane = tid & 31u;
  const unsigned c0   = wave * 16u;   // output column tile
  const unsigned m    = lane & 15u;   // A row within tile
  const unsigned kh   = lane >> 4;    // A K-half select

  const unsigned* Aw = (const unsigned*)As;   // dword view (two bf16 per dword)
  v8f acc = {};

  #pragma unroll
  for (unsigned kb = 0; kb < KTOT; kb += 32) {
    BF16x16 a, b;
    // A fragment: VGPR v holds K = kb + 16*(v>=4) + 8*kh + 2*(v%4)  (pair K,K+1)
    #pragma unroll
    for (unsigned v = 0; v < 8; ++v) {
      unsigned k = kb + ((v & 4u) << 2) + (kh << 3) + ((v & 3u) << 1);
      a.u[v] = Aw[m * (KTOT / 2) + (k >> 1)];
    }
    // B fragment: lane holds col N = c0 + lane%16; VGPR v holds K = kb + 16*(v>=4) + 8*kh + 2*(v%4)  (pair K,K+1)
    const unsigned short* bp = wc + c0 + m;
    #pragma unroll
    for (unsigned v = 0; v < 8; ++v) {
      unsigned k = kb + ((v & 4u) << 2) + (kh << 3) + ((v & 3u) << 1);
      b.u[v] = (unsigned)bp[(size_t)k * 128u] | ((unsigned)bp[(size_t)(k + 1u) * 128u] << 16);
    }

    acc = __builtin_amdgcn_wmma_f32_16x16x32_bf16(
        /*neg_a=*/false, a.v, /*neg_b=*/false, b.v,
        /*c_mod=*/(short)0, acc, /*reuse_a=*/false, /*reuse_b=*/false);
  }

  // C/D layout: VGPR v -> M = v + 8*(lane/16), N = lane%16
  const unsigned n  = lane & 15u;
  const float    bi = bias[c0 + n];
  #pragma unroll
  for (unsigned v = 0; v < 8; ++v) {
    unsigned mrow = v + (kh << 3);
    out[(size_t)(r0 + mrow) * 128u + c0 + n] = 0.5f * acc[v] + bi;
  }
}

extern "C" void kernel_launch(void* const* d_in, const int* in_sizes, int n_in,
                              void* d_out, int out_size, void* d_ws, size_t ws_size,
                              hipStream_t stream) {
  (void)in_sizes; (void)n_in; (void)out_size; (void)ws_size;
  const float* x      = (const float*)d_in[0];
  const int*   rows   = (const int*)  d_in[1];
  const int*   cols   = (const int*)  d_in[2];
  const float* vals   = (const float*)d_in[3];
  const float* weight = (const float*)d_in[4];
  const float* shw    = (const float*)d_in[5];
  const float* bias   = (const float*)d_in[6];
  const float* gamma  = (const float*)d_in[7];
  const float* beta   = (const float*)d_in[8];

  char* ws = (char*)d_ws;
  float* supp = (float*)ws;                               // N*384 f32 = 76.8 MB
  size_t off = (size_t)N_NODES * 384 * sizeof(float);
  unsigned short* normbf = (unsigned short*)(ws + off);   // N*128 bf16 = 12.8 MB
  off += (size_t)N_NODES * 128 * sizeof(unsigned short);
  unsigned short* wc = (unsigned short*)(ws + off);       // 512*128 bf16

  {
    int n4 = N_NODES * 384 / 4;
    k_zero_f32<<<(n4 + 255) / 256, 256, 0, stream>>>(supp, n4);
  }
  {
    long long tw = (long long)R_REL * NNZ_E * 32;   // one wave per edge
    k_spmm<<<(unsigned)((tw + 255) / 256), 256, 0, stream>>>(x, rows, cols, vals, supp);
  }
  k_layernorm<<<(N_NODES + 7) / 8, 256, 0, stream>>>(supp, gamma, beta, normbf);
  {
    int n = KTOT * D;
    k_packw<<<(n + 255) / 256, 256, 0, stream>>>(weight, shw, wc);
  }
  k_gemm<<<N_NODES / 16, 256, 0, stream>>>(supp, normbf, wc, bias, (float*)d_out);
}
